// MyConvolution2D_12687333392736
// MI455X (gfx1250) — matrix operand order repaired and checked by reading, not compiled, not run on hardware
//
#include <hip/hip_runtime.h>

// CDNA5 / gfx1250: wave32, WMMA 16x16x32 bf16, f32 accumulate.
typedef __attribute__((ext_vector_type(16))) __bf16 v16bf;
typedef __attribute__((ext_vector_type(8)))  __bf16 v8bf;
typedef __attribute__((ext_vector_type(8)))  float  v8f;

#define HH   56
#define WW   56
#define CIN  128
#define NOCH 256
#define KTOT 1152              // 9 * 128
#define NCHUNK 36              // KTOT / 32
#define NPIX (32 * HH * WW)    // 100352, M dimension
#define WELEMS (NOCH * KTOT)   // 294912 weight elements
#define MT   32                // pixels per block

// ---------------------------------------------------------------------------
// Pass 1: split fp32 weights into bf16 hi/lo planes (once per launch).
// ---------------------------------------------------------------------------
__global__ __launch_bounds__(512)
void split_weights_bf16(const float* __restrict__ w,
                        __bf16* __restrict__ wH,
                        __bf16* __restrict__ wL)
{
    int i = blockIdx.x * 512 + threadIdx.x;
    if (i < WELEMS) {
        float v = w[i];
        __bf16 h = (__bf16)v;
        wH[i] = h;
        wL[i] = (__bf16)(v - (float)h);
    }
}

// ---------------------------------------------------------------------------
// Pass 2: implicit-GEMM 3x3 SAME conv.
// Block = 32 pixels x 256 out-channels, 512 threads = 16 waves.
// Wave w owns a 32x16 strip (two 16x16 WMMA tiles) for channels [16w,16w+16):
// each B fragment feeds 2 A fragments -> 6 WMMA per K-chunk per wave.
// Double-buffered LDS A-staging (bf16 hi/lo converted once at stage time),
// single barrier per chunk, incremental tap pointers (no in-loop div/mod).
// ---------------------------------------------------------------------------
__global__ __launch_bounds__(512)
void conv3x3_wmma_bf16x3(const float* __restrict__ x,
                         const __bf16* __restrict__ wH,
                         const __bf16* __restrict__ wL,
                         const float* __restrict__ bias,
                         float* __restrict__ out)
{
    // 32 pixels x 32 K bf16, row stride 40 (80 B: 16B-aligned, bank-skewed).
    __shared__ __bf16 lAH[2][MT][40];
    __shared__ __bf16 lAL[2][MT][40];

    const int tid    = threadIdx.x;
    const int lane   = tid & 31;
    const int wave   = tid >> 5;            // 0..15
    const int laneLo = lane & 15;
    const int laneHi = lane >> 4;           // half-wave select

    const int pixBase = blockIdx.x * MT;

    // ---- staging coords: 2 pixels per thread, one float each per chunk ----
    const int sk  = tid & 31;               // k within chunk
    const int sp0 = tid >> 5;               // pixel rows sp0 and sp0+16
    const int sp1 = sp0 + 16;

    // Per-pixel: validity bitmasks over ky/kx and tap-(0,0) base pointer.
    int hm0, wm0, hm1, wm1;
    const float *p0, *p1;
    {
        const int pg0 = pixBase + sp0;
        const int b0  = pg0 / (HH * WW);
        const int r0  = pg0 % (HH * WW);
        const int h0  = r0 / WW, w0 = r0 % WW;
        hm0 = ((h0 >= 1) ? 1 : 0) | 2 | ((h0 < HH - 1) ? 4 : 0);
        wm0 = ((w0 >= 1) ? 1 : 0) | 2 | ((w0 < WW - 1) ? 4 : 0);
        p0  = x + (long)((b0 * HH + h0 - 1) * WW + (w0 - 1)) * CIN + sk;

        const int pg1 = pixBase + sp1;
        const int b1  = pg1 / (HH * WW);
        const int r1  = pg1 % (HH * WW);
        const int h1  = r1 / WW, w1 = r1 % WW;
        hm1 = ((h1 >= 1) ? 1 : 0) | 2 | ((h1 < HH - 1) ? 4 : 0);
        wm1 = ((w1 >= 1) ? 1 : 0) | 2 | ((w1 < WW - 1) ? 4 : 0);
        p1  = x + (long)((b1 * HH + h1 - 1) * WW + (w1 - 1)) * CIN + sk;
    }

    // ---- B operand base: lane -> N, halves hold 16 consecutive K ----
    const int n = wave * 16 + laneLo;       // output channel 0..255
    const __bf16* bHp = wH + (size_t)n * KTOT + laneHi * 16;
    const __bf16* bLp = wL + (size_t)n * KTOT + laneHi * 16;

    // ---- tap walking state ----
    int ky = 0, kx = 0;
    bool ok0 = (hm0 & wm0 & 1) != 0;
    bool ok1 = (hm1 & wm1 & 1) != 0;

    v8f acc0 = {}, acc1 = {};
    float xv0 = ok0 ? p0[0] : 0.0f;
    float xv1 = ok1 ? p1[0] : 0.0f;
    int buf = 0;

    for (int ch = 0; ch < NCHUNK; ++ch) {
        // ---- stage this chunk's A slice into LDS as bf16 hi/lo ----
        {
            __bf16 h0 = (__bf16)xv0;
            lAH[buf][sp0][sk] = h0;
            lAL[buf][sp0][sk] = (__bf16)(xv0 - (float)h0);
            __bf16 h1 = (__bf16)xv1;
            lAH[buf][sp1][sk] = h1;
            lAL[buf][sp1][sk] = (__bf16)(xv1 - (float)h1);
        }
        __syncthreads();

        // ---- prefetch next chunk (constant-stride tap walk, no div/mod) ----
        const int nxt = ch + 1;
        if (nxt < NCHUNK) {
            const int cs = nxt & 3;
            if (cs == 0) {                  // advance to next 3x3 tap
                if (++kx == 3) {
                    kx = 0; ++ky;
                    p0 += (WW - 2) * CIN;
                    p1 += (WW - 2) * CIN;
                } else {
                    p0 += CIN;
                    p1 += CIN;
                }
                ok0 = ((hm0 >> ky) & (wm0 >> kx) & 1) != 0;
                ok1 = ((hm1 >> ky) & (wm1 >> kx) & 1) != 0;
            }
            xv0 = ok0 ? p0[cs * 32] : 0.0f;
            xv1 = ok1 ? p1[cs * 32] : 0.0f;
        }

        // ---- A fragments: lane -> M, halves hold 16 consecutive K ----
        const int r0 = laneLo, r1 = laneLo + 16;
        v16bf aH0 = __builtin_shufflevector(
            *(const v8bf*)&lAH[buf][r0][laneHi * 16],
            *(const v8bf*)&lAH[buf][r0][laneHi * 16 + 8],
            0,1,2,3,4,5,6,7,8,9,10,11,12,13,14,15);
        v16bf aL0 = __builtin_shufflevector(
            *(const v8bf*)&lAL[buf][r0][laneHi * 16],
            *(const v8bf*)&lAL[buf][r0][laneHi * 16 + 8],
            0,1,2,3,4,5,6,7,8,9,10,11,12,13,14,15);
        v16bf aH1 = __builtin_shufflevector(
            *(const v8bf*)&lAH[buf][r1][laneHi * 16],
            *(const v8bf*)&lAH[buf][r1][laneHi * 16 + 8],
            0,1,2,3,4,5,6,7,8,9,10,11,12,13,14,15);
        v16bf aL1 = __builtin_shufflevector(
            *(const v8bf*)&lAL[buf][r1][laneHi * 16],
            *(const v8bf*)&lAL[buf][r1][laneHi * 16 + 8],
            0,1,2,3,4,5,6,7,8,9,10,11,12,13,14,15);

        // ---- B fragments: 16 consecutive pre-split bf16 (32B-aligned) ----
        v16bf bH = *(const v16bf*)(bHp + ch * 32);
        v16bf bL = *(const v16bf*)(bLp + ch * 32);

        // 6 WMMA, two independent accumulator chains interleaved
        acc0 = __builtin_amdgcn_wmma_f32_16x16x32_bf16(false, aH0, false, bH,
                                                       (short)0, acc0, false, false);
        acc1 = __builtin_amdgcn_wmma_f32_16x16x32_bf16(false, aH1, false, bH,
                                                       (short)0, acc1, false, false);
        acc0 = __builtin_amdgcn_wmma_f32_16x16x32_bf16(false, aL0, false, bH,
                                                       (short)0, acc0, false, false);
        acc1 = __builtin_amdgcn_wmma_f32_16x16x32_bf16(false, aL1, false, bH,
                                                       (short)0, acc1, false, false);
        acc0 = __builtin_amdgcn_wmma_f32_16x16x32_bf16(false, aH0, false, bL,
                                                       (short)0, acc0, false, false);
        acc1 = __builtin_amdgcn_wmma_f32_16x16x32_bf16(false, aH1, false, bL,
                                                       (short)0, acc1, false, false);
        buf ^= 1;
    }

    // ---- epilogue: C/D layout lane -> N, vgpr v -> M = v + 8*laneHi ----
    const float bv = bias[n];
    #pragma unroll
    for (int v = 0; v < 8; ++v) {
        const int m = v + laneHi * 8;
        out[(size_t)(pixBase + m) * NOCH + n]      = acc0[v] + bv;
        out[(size_t)(pixBase + 16 + m) * NOCH + n] = acc1[v] + bv;
    }
}

// ---------------------------------------------------------------------------
// Fallback (ws too small): monolithic kernel with in-loop conversion.
// ---------------------------------------------------------------------------
__global__ __launch_bounds__(512)
void conv3x3_wmma_fallback(const float* __restrict__ x,
                           const float* __restrict__ w,
                           const float* __restrict__ bias,
                           float* __restrict__ out)
{
    __shared__ float lA[16][36];

    const int tid    = threadIdx.x;
    const int lane   = tid & 31;
    const int wave   = tid >> 5;
    const int laneLo = lane & 15;
    const int laneHi = lane >> 4;
    const int pixBase = blockIdx.x * 16;

    const int sp = tid >> 5;
    const int sk = tid & 31;
    const int pg = pixBase + sp;
    const int bb = pg / (HH * WW);
    const int hw = pg % (HH * WW);
    const int hh = hw / WW;
    const int wc = hw % WW;

    const int n = wave * 16 + laneLo;
    const float* wrow = w + (size_t)n * KTOT + laneHi * 16;

    v8f acc = {};

    for (int ch = 0; ch < NCHUNK; ++ch) {
        const int kk = ch >> 2;
        const int ky = kk / 3;
        const int kx = kk % 3;
        const int c0 = (ch & 3) * 32;

        __syncthreads();
        {
            const int hs = hh + ky - 1;
            const int ws = wc + kx - 1;
            float v = 0.0f;
            if (hs >= 0 && hs < HH && ws >= 0 && ws < WW)
                v = x[((size_t)((bb * HH + hs) * WW + ws)) * CIN + c0 + sk];
            lA[sp][sk] = v;
        }
        __syncthreads();

        const float* arow = &lA[laneLo][0];
        float af[16];
        #pragma unroll
        for (int i = 0; i < 8; ++i) af[i]     = arow[laneHi * 16 + i];
        #pragma unroll
        for (int i = 0; i < 8; ++i) af[8 + i] = arow[laneHi * 16 + 8 + i];

        const float* bptr = wrow + ch * 32;
        float bf[16];
        #pragma unroll
        for (int i = 0; i < 16; ++i) bf[i] = bptr[i];

        v16bf aH, aL, bH, bL;
        #pragma unroll
        for (int i = 0; i < 16; ++i) {
            __bf16 h = (__bf16)af[i];
            aH[i] = h;
            aL[i] = (__bf16)(af[i] - (float)h);
            __bf16 g = (__bf16)bf[i];
            bH[i] = g;
            bL[i] = (__bf16)(bf[i] - (float)g);
        }

        acc = __builtin_amdgcn_wmma_f32_16x16x32_bf16(false, aH, false, bH,
                                                      (short)0, acc, false, false);
        acc = __builtin_amdgcn_wmma_f32_16x16x32_bf16(false, aL, false, bH,
                                                      (short)0, acc, false, false);
        acc = __builtin_amdgcn_wmma_f32_16x16x32_bf16(false, aH, false, bL,
                                                      (short)0, acc, false, false);
    }

    const float bv = bias[n];
    #pragma unroll
    for (int v = 0; v < 8; ++v) {
        const int m = v + laneHi * 8;
        out[(size_t)(pixBase + m) * NOCH + n] = acc[v] + bv;
    }
}

extern "C" void kernel_launch(void* const* d_in, const int* in_sizes, int n_in,
                              void* d_out, int out_size, void* d_ws, size_t ws_size,
                              hipStream_t stream)
{
    const float* x  = (const float*)d_in[0];
    const float* w  = (const float*)d_in[1];
    const float* b  = (const float*)d_in[2];
    float* out = (float*)d_out;

    const size_t wsNeeded = (size_t)2 * WELEMS * sizeof(__bf16);  // ~1.13 MB

    if (ws_size >= wsNeeded && d_ws != nullptr) {
        __bf16* wHp = (__bf16*)d_ws;
        __bf16* wLp = wHp + WELEMS;
        split_weights_bf16<<<(WELEMS + 511) / 512, 512, 0, stream>>>(w, wHp, wLp);
        conv3x3_wmma_bf16x3<<<NPIX / MT, 512, 0, stream>>>(x, wHp, wLp, b, out);
    } else {
        conv3x3_wmma_fallback<<<NPIX / 16, 512, 0, stream>>>(x, w, b, out);
    }
}
